// GuidanceConstruct_46823733461711
// MI455X (gfx1250) — matrix operand order repaired and checked by reading, not compiled, not run on hardware
//
#include <hip/hip_runtime.h>
#include <hip/hip_bf16.h>

// Problem constants (from setup_inputs)
#define Bq   8
#define Cq   64
#define Kq   64
#define Lq   128
#define TOT  (Bq*Cq*Kq*Lq)    // 4,194,304
#define BLn  (Bq*Lq)          // 1024
#define BKn  (Bq*Kq)          // 512

typedef __attribute__((ext_vector_type(16))) __bf16 v16bf;
typedef __attribute__((ext_vector_type(8)))  float  v8f;

// ---------------------------------------------------------------------------
// Generic bf16-WMMA GEMM: Out[m,n] = epi(alpha * sum_k A[m,k]*B(k,n))
//   BTR==1: B(k,n) = Bm[n*ldb + k]  (weight stored [N,K] row-major, i.e. X @ W^T)
//   BTR==0: B(k,n) = Bm[k*ldb + n]  (plain row-major [K,N])
// EPI (compile-time epilogue):
//   0: + bias[m] (row bias), store
//   1: store
//   2: Out += v (accumulate)
//   3: + bias[n], silu
//   4: + bias[n] + Res, store
//   5: + bias[n], store
//   6: + bias[n], gelu(erf)
//   7: + bias[n], relu
// One wave computes a 16x32 output block (two WMMAs sharing the A fragment);
// 4 waves per block; batch via grid.y.
// ---------------------------------------------------------------------------
template <int BTR, int EPI>
__global__ __launch_bounds__(128)
void gemm_wmma_k(const float* __restrict__ A, int lda, long sA,
                 const float* __restrict__ Bm, int ldb, long sB,
                 const float* __restrict__ bias,
                 const float* __restrict__ Res,
                 float* __restrict__ Out, int ldo, long sO,
                 int M, int N, int Kd, float alpha)
{
    int wave = threadIdx.x >> 5;
    int lane = threadIdx.x & 31;
    int tilesN = N >> 5;                      // 32-wide N blocks
    int tiles  = (M >> 4) * tilesN;
    int tile   = blockIdx.x * 4 + wave;       // wave-uniform
    if (tile >= tiles) return;                // whole-wave exit: EXEC stays full for WMMA
    int tm = tile / tilesN;
    int tn = tile - tm * tilesN;
    long batch = blockIdx.y;
    A   += batch * sA;
    Bm  += batch * sB;
    Out += batch * sO;
    const float* Rp = (EPI == 4) ? (Res + batch * sO) : (const float*)0;

    int half = lane >> 4;      // 0: lanes 0-15, 1: lanes 16-31
    int l15  = lane & 15;

    v8f acc0 = {0.f,0.f,0.f,0.f,0.f,0.f,0.f,0.f};
    v8f acc1 = {0.f,0.f,0.f,0.f,0.f,0.f,0.f,0.f};
    for (int kc = 0; kc < Kd; kc += 32) {
        v16bf av, bv0, bv1;
        // A 16x32 bf16 layout: lane m=l15; VGPR i holds K pair at
        //   kbase = half*16 + i*2
        const float* arow = A + (size_t)(tm*16 + l15) * lda + kc;
#pragma unroll
        for (int i = 0; i < 8; ++i) {
            int kb = (half << 4) + (i << 1);
            av[2*i]   = (__bf16)arow[kb];
            av[2*i+1] = (__bf16)arow[kb + 1];
        }
        // B 32x16 bf16 layout: lane n=l15; lanes 0-15 hold K=0..15 (2/VGPR),
        // lanes 16-31 hold K=16..31.
        if (BTR) {
            const float* brow0 = Bm + (size_t)(tn*32 + l15)      * ldb + kc + (half << 4);
            const float* brow1 = Bm + (size_t)(tn*32 + 16 + l15) * ldb + kc + (half << 4);
#pragma unroll
            for (int i = 0; i < 8; ++i) {
                bv0[2*i]   = (__bf16)brow0[2*i];
                bv0[2*i+1] = (__bf16)brow0[2*i + 1];
                bv1[2*i]   = (__bf16)brow1[2*i];
                bv1[2*i+1] = (__bf16)brow1[2*i + 1];
            }
        } else {
            const float* bcol = Bm + (size_t)(kc + (half << 4)) * ldb + tn*32 + l15;
#pragma unroll
            for (int i = 0; i < 8; ++i) {
                bv0[2*i]   = (__bf16)bcol[(size_t)(2*i)     * ldb];
                bv0[2*i+1] = (__bf16)bcol[(size_t)(2*i + 1) * ldb];
                bv1[2*i]   = (__bf16)bcol[(size_t)(2*i)     * ldb + 16];
                bv1[2*i+1] = (__bf16)bcol[(size_t)(2*i + 1) * ldb + 16];
            }
        }
        acc0 = __builtin_amdgcn_wmma_f32_16x16x32_bf16(false, av, false, bv0,
                                                       (short)0, acc0, false, false);
        acc1 = __builtin_amdgcn_wmma_f32_16x16x32_bf16(false, av, false, bv1,
                                                       (short)0, acc1, false, false);
    }

    // D layout: VGPR r -> row m = tm*16 + half*8 + r, col n = tn*32 (+16) + l15
#pragma unroll
    for (int t = 0; t < 2; ++t) {
        int n = tn*32 + t*16 + l15;
        float bn = 0.f;
        if (EPI >= 3) bn = bias[n];            // column bias, hoisted
#pragma unroll
        for (int r = 0; r < 8; ++r) {
            int m = tm*16 + (half << 3) + r;
            float v = (t == 0 ? acc0[r] : acc1[r]) * alpha;
            size_t oi = (size_t)m * ldo + n;
            if (EPI == 0) v += bias[m];
            if (EPI >= 3) v += bn;
            if (EPI == 4) v += Rp[oi];
            if (EPI == 2) v += Out[oi];
            if (EPI == 3) v = v / (1.f + __expf(-v));                            // silu
            if (EPI == 6) v = 0.5f * v * (1.f + erff(v * 0.70710678118654752f)); // gelu exact
            if (EPI == 7) v = fmaxf(v, 0.f);                                     // relu
            Out[oi] = v;
        }
    }
}

// ---------------------------------------------------------------------------
// Layout shuffles. y is [B, C, K*L] with tid = ((b*64+c)*64+k)*128 + l.
// ---------------------------------------------------------------------------
__global__ void spread_k(const float* __restrict__ y,
                         float* __restrict__ xs, float* __restrict__ xsp)
{
    int tid = blockIdx.x * 256 + threadIdx.x;
    if (tid >= TOT) return;
    int l = tid & 127, k = (tid >> 7) & 63, c = (tid >> 13) & 63, b = tid >> 19;
    float v = y[tid];
    int n = b * Lq + l;
    xs [((size_t)n * 64 + c) * 64 + k] = v;   // [BL, C, K]
    xsp[((size_t)n * 64 + k) * 64 + c] = v;   // [BL, K, C]
}

__global__ void scatter_xt_k(const float* __restrict__ y, float* __restrict__ xt)
{
    int tid = blockIdx.x * 256 + threadIdx.x;
    if (tid >= TOT) return;
    int l = tid & 127, k = (tid >> 7) & 63, c = (tid >> 13) & 63, b = tid >> 19;
    xt[((size_t)(l * BKn + b * Kq + k)) * 64 + c] = y[tid];   // rows (l, n=b*K+k)
}

__global__ void scatter_tbuf_k(const float* __restrict__ yin2, float* __restrict__ tb)
{
    int tid = blockIdx.x * 256 + threadIdx.x;
    if (tid >= TOT) return;
    int l = tid & 127, k = (tid >> 7) & 63, c = (tid >> 13) & 63, b = tid >> 19;
    tb[((size_t)(b * 8192 + k * 128 + l)) * 64 + c] = yin2[tid];  // rows (b, s)
}

// assemble back into [B, C, K*L] and add `add` (the y residual / y_in2)
// mode 0: src [n, c, k]; 1: src [n, k, c]; 2: src [(l*BK+n), c]; 3: src [(b*KL+s), c]
__global__ void assemble_k(const float* __restrict__ src, const float* __restrict__ add,
                           float* __restrict__ out, int mode)
{
    int tid = blockIdx.x * 256 + threadIdx.x;
    if (tid >= TOT) return;
    int l = tid & 127, k = (tid >> 7) & 63, c = (tid >> 13) & 63, b = tid >> 19;
    size_t idx;
    if (mode == 0)      idx = ((size_t)(b * Lq + l) * 64 + c) * 64 + k;
    else if (mode == 1) idx = ((size_t)(b * Lq + l) * 64 + k) * 64 + c;
    else if (mode == 2) idx =  (size_t)(l * BKn + b * Kq + k) * 64 + c;
    else                idx =  (size_t)(b * 8192 + k * 128 + l) * 64 + c;
    out[tid] = src[idx] + add[tid];
}

__global__ void sum3_k(const float* __restrict__ a, const float* __restrict__ b,
                       const float* __restrict__ c, float* __restrict__ o)
{
    int tid = blockIdx.x * 256 + threadIdx.x;
    if (tid >= TOT) return;
    o[tid] = a[tid] + b[tid] + c[tid];
}

// ---------------------------------------------------------------------------
// GroupNorm over [B, 64, 8192], groups=4 (16 ch x 8192 = 131072 elems / group)
// ---------------------------------------------------------------------------
__global__ void groupnorm_k(const float* __restrict__ in, const float* __restrict__ w,
                            const float* __restrict__ bias, float* __restrict__ out)
{
    int bg = blockIdx.x;               // 0..31
    int b = bg >> 2, g = bg & 3;
    const float* base = in  + ((size_t)b * 64 + g * 16) * 8192;
    float*       ob   = out + ((size_t)b * 64 + g * 16) * 8192;
    float s = 0.f, ss = 0.f;
    for (int i = threadIdx.x; i < 131072; i += 256) { float v = base[i]; s += v; ss += v * v; }
    __shared__ float rs[256], rq[256];
    rs[threadIdx.x] = s; rq[threadIdx.x] = ss; __syncthreads();
    for (int st = 128; st > 0; st >>= 1) {
        if (threadIdx.x < st) { rs[threadIdx.x] += rs[threadIdx.x + st];
                                rq[threadIdx.x] += rq[threadIdx.x + st]; }
        __syncthreads();
    }
    float mean = rs[0] * (1.f / 131072.f);
    float var  = rq[0] * (1.f / 131072.f) - mean * mean;
    float inv  = rsqrtf(var + 1e-5f);
    for (int i = threadIdx.x; i < 131072; i += 256) {
        int c = g * 16 + (i >> 13);
        ob[i] = (base[i] - mean) * inv * w[c] + bias[c];
    }
}

// LayerNorm over last dim 64; one block (64 threads) per row
__global__ void layernorm_k(const float* __restrict__ in, const float* __restrict__ w,
                            const float* __restrict__ bias, float* __restrict__ out)
{
    int r = blockIdx.x, t = threadIdx.x;
    float v = in[(size_t)r * 64 + t];
    __shared__ float s1[64], s2[64];
    s1[t] = v; s2[t] = v * v; __syncthreads();
    for (int st = 32; st > 0; st >>= 1) {
        if (t < st) { s1[t] += s1[t + st]; s2[t] += s2[t + st]; }
        __syncthreads();
    }
    float m   = s1[0] * (1.f / 64.f);
    float var = s2[0] * (1.f / 64.f) - m * m;
    float inv = rsqrtf(var + 1e-5f);
    out[(size_t)r * 64 + t] = (v - m) * inv * w[t] + bias[t];
}

// adp = softmax_rows(relu(nv1 @ nv2)), K=64
__global__ void adp_k(const float* __restrict__ nv1, const float* __restrict__ nv2,
                      float* __restrict__ adp)
{
    int i = blockIdx.x, j = threadIdx.x;   // 64 x 64
    float acc = 0.f;
    for (int t = 0; t < 10; ++t) acc += nv1[i * 10 + t] * nv2[t * 64 + j];
    acc = fmaxf(acc, 0.f);
    __shared__ float sm[64];
    sm[j] = acc; __syncthreads();
    for (int st = 32; st > 0; st >>= 1) { if (j < st) sm[j] = fmaxf(sm[j], sm[j + st]); __syncthreads(); }
    float mx = sm[0]; __syncthreads();
    float e = __expf(acc - mx);
    sm[j] = e; __syncthreads();
    for (int st = 32; st > 0; st >>= 1) { if (j < st) sm[j] += sm[j + st]; __syncthreads(); }
    adp[i * 64 + j] = e / sm[0];
}

// ---------------------------------------------------------------------------
// Fused spatial attention (per (bl, head)): online softmax over 64 nodes.
// attn = softmax(pi*logits + q.kT/sqrt(8)); o = attn @ v
// ---------------------------------------------------------------------------
__global__ void spa_attn_k(const float* __restrict__ qb, const float* __restrict__ kb,
                           const float* __restrict__ vb, const float* __restrict__ lg,
                           const float* __restrict__ pi_p, float* __restrict__ ob)
{
    int n = blockIdx.x, h = blockIdx.y, i = threadIdx.x;   // i = query node 0..63
    float pi = pi_p[0];
    __shared__ float kh[64][8], vh[64][8];
    for (int e = threadIdx.x; e < 512; e += 64) {
        int j = e >> 3, d = e & 7;
        kh[j][d] = kb[(size_t)n * 4096 + j * 64 + h * 8 + d];
        vh[j][d] = vb[(size_t)n * 4096 + j * 64 + h * 8 + d];
    }
    __syncthreads();
    float qr[8];
#pragma unroll
    for (int d = 0; d < 8; ++d) qr[d] = qb[(size_t)n * 4096 + i * 64 + h * 8 + d];
    const float* lrow = lg + (size_t)n * 4096 + i * 64;
    const float scale = 0.35355339059327373f;  // 1/sqrt(8)
    float mx = -1e30f, sum = 0.f, acc[8] = {0,0,0,0,0,0,0,0};
    for (int j = 0; j < 64; ++j) {
        float s = 0.f;
#pragma unroll
        for (int d = 0; d < 8; ++d) s += qr[d] * kh[j][d];
        s = s * scale + pi * lrow[j];
        float m2 = fmaxf(mx, s);
        float corr = __expf(mx - m2);
        float p = __expf(s - m2);
        sum = sum * corr + p;
#pragma unroll
        for (int d = 0; d < 8; ++d) acc[d] = acc[d] * corr + p * vh[j][d];
        mx = m2;
    }
    float inv = 1.f / sum;
#pragma unroll
    for (int d = 0; d < 8; ++d)
        ob[(size_t)n * 4096 + i * 64 + h * 8 + d] = acc[d] * inv;
}

// Fused temporal attention (per (n, head)): online softmax over 128 steps.
// qkv rows (l*BK + n), cols [q(0:64) k(64:128) v(128:192)]
__global__ void tem_attn_k(const float* __restrict__ qkv, float* __restrict__ ao)
{
    int n = blockIdx.x, h = blockIdx.y, i = threadIdx.x;   // i = query time 0..127
    __shared__ float kh[128][8], vh[128][8];
    for (int e = threadIdx.x; e < 1024; e += 128) {
        int j = e >> 3, d = e & 7;
        size_t row = (size_t)j * BKn + n;
        kh[j][d] = qkv[row * 192 + 64  + h * 8 + d];
        vh[j][d] = qkv[row * 192 + 128 + h * 8 + d];
    }
    __syncthreads();
    size_t qrow = (size_t)i * BKn + n;
    float qr[8];
#pragma unroll
    for (int d = 0; d < 8; ++d) qr[d] = qkv[qrow * 192 + h * 8 + d];
    const float scale = 0.35355339059327373f;
    float mx = -1e30f, sum = 0.f, acc[8] = {0,0,0,0,0,0,0,0};
    for (int j = 0; j < 128; ++j) {
        float s = 0.f;
#pragma unroll
        for (int d = 0; d < 8; ++d) s += qr[d] * kh[j][d];
        s *= scale;
        float m2 = fmaxf(mx, s);
        float corr = __expf(mx - m2);
        float p = __expf(s - m2);
        sum = sum * corr + p;
#pragma unroll
        for (int d = 0; d < 8; ++d) acc[d] = acc[d] * corr + p * vh[j][d];
        mx = m2;
    }
    float inv = 1.f / sum;
#pragma unroll
    for (int d = 0; d < 8; ++d)
        ao[qrow * 64 + h * 8 + d] = acc[d] * inv;
}

// ---------------------------------------------------------------------------
// Host orchestration
// ---------------------------------------------------------------------------
template <int BTR>
static inline void launch_gemm_t(hipStream_t st, int epi,
                                 const float* A, int lda, long sA,
                                 const float* Bm, int ldb, long sB,
                                 const float* bias, const float* Res,
                                 float* Out, int ldo, long sO,
                                 int M, int N, int Kd, float alpha, int batch)
{
    int tiles = (M / 16) * (N / 32);
    dim3 grid((tiles + 3) / 4, batch);
    dim3 blk(128);
#define GEMM_CASE(E) case E: \
    gemm_wmma_k<BTR, E><<<grid, blk, 0, st>>>(A, lda, sA, Bm, ldb, sB, bias, Res, \
                                              Out, ldo, sO, M, N, Kd, alpha); break;
    switch (epi) {
        GEMM_CASE(0) GEMM_CASE(1) GEMM_CASE(2) GEMM_CASE(3)
        GEMM_CASE(4) GEMM_CASE(5) GEMM_CASE(6) GEMM_CASE(7)
        default: break;
    }
#undef GEMM_CASE
}

static inline void launch_gemm(hipStream_t st, int btr, int epi,
                               const float* A, int lda, long sA,
                               const float* Bm, int ldb, long sB,
                               const float* bias, const float* Res,
                               float* Out, int ldo, long sO,
                               int M, int N, int Kd, float alpha, int batch)
{
    if (btr) launch_gemm_t<1>(st, epi, A, lda, sA, Bm, ldb, sB, bias, Res, Out, ldo, sO, M, N, Kd, alpha, batch);
    else     launch_gemm_t<0>(st, epi, A, lda, sA, Bm, ldb, sB, bias, Res, Out, ldo, sO, M, N, Kd, alpha, batch);
}

extern "C" void kernel_launch(void* const* d_in, const int* in_sizes, int n_in,
                              void* d_out, int out_size, void* d_ws, size_t ws_size,
                              hipStream_t stream)
{
    (void)in_sizes; (void)n_in; (void)out_size; (void)ws_size;
    const float* y          = (const float*)d_in[0];
    const float* a1         = (const float*)d_in[1];
    const float* a2         = (const float*)d_in[2];
    const float* nodevec1   = (const float*)d_in[3];
    const float* nodevec2   = (const float*)d_in[4];
    const float* gcn_w      = (const float*)d_in[5];   // [64, 448]
    const float* gcn_b      = (const float*)d_in[6];
    const float* pre_w      = (const float*)d_in[7];
    const float* pre_b      = (const float*)d_in[8];
    const float* pi         = (const float*)d_in[9];
    const float* q_w        = (const float*)d_in[10];
    const float* k_w        = (const float*)d_in[11];
    const float* v_w        = (const float*)d_in[12];
    const float* proj_w     = (const float*)d_in[13];
    const float* proj_b     = (const float*)d_in[14];
    const float* in_proj_w  = (const float*)d_in[15];  // [192, 64]
    const float* in_proj_b  = (const float*)d_in[16];
    const float* out_proj_w = (const float*)d_in[17];
    const float* out_proj_b = (const float*)d_in[18];
    const float* lin1_w     = (const float*)d_in[19];
    const float* lin1_b     = (const float*)d_in[20];
    const float* lin2_w     = (const float*)d_in[21];
    const float* lin2_b     = (const float*)d_in[22];
    const float* ln1_w      = (const float*)d_in[23];
    const float* ln1_b      = (const float*)d_in[24];
    const float* ln2_w      = (const float*)d_in[25];
    const float* ln2_b      = (const float*)d_in[26];
    const float* gn_local_w = (const float*)d_in[27];
    const float* gn_local_b = (const float*)d_in[28];
    const float* gn_s_w     = (const float*)d_in[29];
    const float* gn_s_b     = (const float*)d_in[30];
    const float* gn_t_w     = (const float*)d_in[31];
    const float* gn_t_b     = (const float*)d_in[32];
    const float* ff1_w      = (const float*)d_in[33];  // [128, 64]
    const float* ff1_b      = (const float*)d_in[34];
    const float* ff2_w      = (const float*)d_in[35];  // [64, 128]
    const float* ff2_b      = (const float*)d_in[36];
    const float* gn2_w      = (const float*)d_in[37];
    const float* gn2_b      = (const float*)d_in[38];

    // Workspace arena: 12 slots of TOT floats + adp (lifetime-checked aliasing)
    float* WS = (float*)d_ws;
    const size_t S = (size_t)TOT;
    float* xs    = WS + 0 * S;
    float* tmp1  = WS + 1 * S;
    float* tmp2  = WS + 2 * S;
    float* g     = WS + 3 * S;
    float* yloc  = WS + 4 * S;
    float* xsp   = WS + 5 * S;
    float* qbuf  = WS + 6 * S;
    float* kbuf  = WS + 7 * S;
    float* vbuf  = WS + 8 * S;
    float* yatt  = WS + 9 * S;
    float* h1    = WS + 10 * S;          // spans slots 10-11 (65536 x 128)
    float* adpb  = WS + 12 * S;          // 4096 floats
    // aliases (non-overlapping in time)
    float* pre   = tmp1;                 // until logits computed
    float* logit = tmp2;                 // until fused spatial attn
    float* obuf  = tmp1;                 // spatial attn output
    float* sbuf  = g;                    // spatial proj output
    float* xt    = xs;                   // temporal input rows
    float* qkv   = tmp1;                 // spans tmp1..g (65536 x 192)
    float* aobuf = qbuf;                 // temporal attn output
    float* srcb  = tmp2;                 // after qkv dead
    float* ffb   = g;
    float* src2  = xsp;                  // after xsp dead
    float* ytb   = qbuf;                 // after aobuf dead
    float* yin2  = kbuf;
    float* tbuf  = xs;                   // after xt dead
    float* h2    = vbuf;
    float* outp  = tmp1;

    const int eb = (TOT + 255) / 256;

    // ---- Stage 0: layout spreads + adaptive adjacency ----
    spread_k<<<eb, 256, 0, stream>>>(y, xs, xsp);
    adp_k<<<64, 64, 0, stream>>>(nodevec1, nodevec2, adpb);

    // ---- GCN branch: g[n] = sum_j W_j @ X_j[n]  (batched 1024 x [64x64x64]) ----
    // epi 0: row bias; epi 2: accumulate into g
    launch_gemm(stream, 0, 0, gcn_w + 0 * 64, 448, 0, xs, 64, 4096, gcn_b, nullptr,
                g, 64, 4096, 64, 64, 64, 1.f, BLn);
    const float* adjs[3] = {a1, a2, adpb};
    for (int ai = 0; ai < 3; ++ai) {
        // x1 = x @ adj^T  (flat rows (n,c))
        launch_gemm(stream, 1, 1, xs, 64, 0, adjs[ai], 64, 0, nullptr, nullptr,
                    tmp1, 64, 0, BLn * 64, 64, 64, 1.f, 1);
        launch_gemm(stream, 0, 2, gcn_w + (1 + 2 * ai) * 64, 448, 0, tmp1, 64, 4096,
                    nullptr, nullptr, g, 64, 4096, 64, 64, 64, 1.f, BLn);
        launch_gemm(stream, 1, 1, tmp1, 64, 0, adjs[ai], 64, 0, nullptr, nullptr,
                    tmp2, 64, 0, BLn * 64, 64, 64, 1.f, 1);
        launch_gemm(stream, 0, 2, gcn_w + (2 + 2 * ai) * 64, 448, 0, tmp2, 64, 4096,
                    nullptr, nullptr, g, 64, 4096, 64, 64, 64, 1.f, BLn);
    }
    assemble_k<<<eb, 256, 0, stream>>>(g, y, yloc, 0);
    groupnorm_k<<<32, 256, 0, stream>>>(yloc, gn_local_w, gn_local_b, yloc);

    // ---- Spatial attention branch ----
    launch_gemm(stream, 1, 3 /*bias+silu*/, xsp, 64, 0, pre_w, 64, 0, pre_b, nullptr,
                pre, 64, 0, BLn * 64, 64, 64, 1.f, 1);
    launch_gemm(stream, 1, 1, pre, 64, 4096, pre, 64, 4096, nullptr, nullptr,
                logit, 64, 4096, 64, 64, 64, 0.125f /*1/sqrt(C)*/, BLn);
    launch_gemm(stream, 1, 1, xsp, 64, 0, q_w, 64, 0, nullptr, nullptr,
                qbuf, 64, 0, BLn * 64, 64, 64, 1.f, 1);
    launch_gemm(stream, 1, 1, xsp, 64, 0, k_w, 64, 0, nullptr, nullptr,
                kbuf, 64, 0, BLn * 64, 64, 64, 1.f, 1);
    launch_gemm(stream, 1, 1, xsp, 64, 0, v_w, 64, 0, nullptr, nullptr,
                vbuf, 64, 0, BLn * 64, 64, 64, 1.f, 1);
    spa_attn_k<<<dim3(BLn, 8), 64, 0, stream>>>(qbuf, kbuf, vbuf, logit, pi, obuf);
    launch_gemm(stream, 1, 4 /*bias+res*/, obuf, 64, 0, proj_w, 64, 0, proj_b, xsp,
                sbuf, 64, 0, BLn * 64, 64, 64, 1.f, 1);
    assemble_k<<<eb, 256, 0, stream>>>(sbuf, y, yatt, 1);
    groupnorm_k<<<32, 256, 0, stream>>>(yatt, gn_s_w, gn_s_b, yatt);

    // ---- Temporal transformer branch ----
    scatter_xt_k<<<eb, 256, 0, stream>>>(y, xt);
    launch_gemm(stream, 1, 5 /*bias*/, xt, 64, 0, in_proj_w, 64, 0, in_proj_b, nullptr,
                qkv, 192, 0, Lq * BKn, 192, 64, 1.f, 1);
    tem_attn_k<<<dim3(BKn, 8), 128, 0, stream>>>(qkv, aobuf);
    launch_gemm(stream, 1, 4 /*bias+res*/, aobuf, 64, 0, out_proj_w, 64, 0, out_proj_b, xt,
                srcb, 64, 0, Lq * BKn, 64, 64, 1.f, 1);
    layernorm_k<<<Lq * BKn, 64, 0, stream>>>(srcb, ln1_w, ln1_b, srcb);
    launch_gemm(stream, 1, 6 /*bias+gelu*/, srcb, 64, 0, lin1_w, 64, 0, lin1_b, nullptr,
                ffb, 64, 0, Lq * BKn, 64, 64, 1.f, 1);
    launch_gemm(stream, 1, 4 /*bias+res*/, ffb, 64, 0, lin2_w, 64, 0, lin2_b, srcb,
                src2, 64, 0, Lq * BKn, 64, 64, 1.f, 1);
    layernorm_k<<<Lq * BKn, 64, 0, stream>>>(src2, ln2_w, ln2_b, src2);
    assemble_k<<<eb, 256, 0, stream>>>(src2, y, ytb, 2);
    groupnorm_k<<<32, 256, 0, stream>>>(ytb, gn_t_w, gn_t_b, ytb);

    // ---- Fuse + channel FFN ----
    sum3_k<<<eb, 256, 0, stream>>>(yloc, yatt, ytb, yin2);
    scatter_tbuf_k<<<eb, 256, 0, stream>>>(yin2, tbuf);
    launch_gemm(stream, 1, 7 /*bias+relu*/, tbuf, 64, 0, ff1_w, 64, 0, ff1_b, nullptr,
                h1, 128, 0, Bq * 8192, 128, 64, 1.f, 1);
    launch_gemm(stream, 1, 5 /*bias*/, h1, 128, 0, ff2_w, 128, 0, ff2_b, nullptr,
                h2, 64, 0, Bq * 8192, 64, 128, 1.f, 1);
    assemble_k<<<eb, 256, 0, stream>>>(h2, yin2, outp, 3);
    groupnorm_k<<<32, 256, 0, stream>>>(outp, gn2_w, gn2_b, (float*)d_out);
}
